// TeleportationAttention_49778670960941
// MI455X (gfx1250) — matrix operand order repaired and checked by reading, not compiled, not run on hardware
//
#include <hip/hip_runtime.h>
#include <hip/hip_bf16.h>

// ---------------------------------------------------------------------------
// TeleportationAttention for MI455X (gfx1250, wave32, WMMA)
//   Q,K,V = complex Linear(x);  scores = Re(Q K^H);  attn = softmax(beta*scores)
//   out = attn @ V  (complex64 interleaved)
// All GEMMs on v_wmma_f32_16x16x32_f16 (f32 accumulate), 2x2 register blocking
// per wave (32x32 output tile), k-loops unrolled 4x so fragment addresses fold
// into immediate offsets. Two-pass softmax: the 64MB f32 score tensor is
// L2-resident (192MB L2), so materializing it costs ~no HBM traffic.
// ---------------------------------------------------------------------------

typedef _Float16 v16h __attribute__((ext_vector_type(16)));
typedef float    v8f  __attribute__((ext_vector_type(8)));

constexpr int   Bn   = 4;
constexpr int   Sn   = 2048;
constexpr int   Dn   = 512;
constexpr int   Mn   = Bn * Sn;          // 8192 rows total
constexpr float BETA = 1.0f;

#define DEV static __device__ __forceinline__

union FragU { uint4 q[2]; v16h h; };

// A-matrix 16x32 f16 fragment (ISA 7.12.2): lane row m=lane%16; K runs
// [kk+8*half, +8) and [kk+16+8*half, +8)  -> two 16B loads.
DEV v16h loadA(const _Float16* __restrict__ base, int ld, int row, int kk, int lane) {
  const int half = lane >> 4;
  const _Float16* p = base + (size_t)row * ld + kk + 8 * half;
  FragU f;
  f.q[0] = *reinterpret_cast<const uint4*>(p);
  f.q[1] = *reinterpret_cast<const uint4*>(p + 16);
  return f.h;
}

// B-matrix 32x16 f16 fragment: lane column n=lane%16; K runs
// [kk+8*half, +8) and [kk+16+8*half, +8)  -> two 16B loads.
// `row` is the storage row holding column n's K-axis contiguously.
DEV v16h loadB(const _Float16* __restrict__ base, int ld, int row, int kk, int lane) {
  const int half = lane >> 4;
  const _Float16* p = base + (size_t)row * ld + kk + 8 * half;
  FragU f;
  f.q[0] = *reinterpret_cast<const uint4*>(p);
  f.q[1] = *reinterpret_cast<const uint4*>(p + 16);
  return f.h;
}

DEV v16h negH(v16h x) {  // negate all 16 f16 lanes: flip sign bits
  FragU f; f.h = x;
  f.q[0].x ^= 0x80008000u; f.q[0].y ^= 0x80008000u;
  f.q[0].z ^= 0x80008000u; f.q[0].w ^= 0x80008000u;
  f.q[1].x ^= 0x80008000u; f.q[1].y ^= 0x80008000u;
  f.q[1].z ^= 0x80008000u; f.q[1].w ^= 0x80008000u;
  return f.h;
}

DEV v8f wmma16(v16h a, v16h b, v8f c) {
  return __builtin_amdgcn_wmma_f32_16x16x32_f16(
      /*neg_a=*/false, a, /*neg_b=*/false, b,
      /*c_mod=*/(short)0, c, /*reuse_a=*/false, /*reuse_b=*/false);
}

// ---------------------------------------------------------------------------
// f32 -> f16 conversion, 4-wide
// ---------------------------------------------------------------------------
__global__ void cvt_f32_f16_kernel(const float* __restrict__ src,
                                   _Float16* __restrict__ dst, int n4) {
  int i = blockIdx.x * blockDim.x + threadIdx.x;
  if (i >= n4) return;
  float4 v = reinterpret_cast<const float4*>(src)[i];
  union { _Float16 h[4]; uint2 u; } o;
  o.h[0] = (_Float16)v.x; o.h[1] = (_Float16)v.y;
  o.h[2] = (_Float16)v.z; o.h[3] = (_Float16)v.w;
  reinterpret_cast<uint2*>(dst)[i] = o.u;
}

// ---------------------------------------------------------------------------
// Complex projection:  out = x @ W^T + b   (one of Q/K/V)
//   out_re = xr@Wre^T - xi@Wim^T + bre ;  out_im = xr@Wim^T + xi@Wre^T + bim
// 2x2 register blocking: wave computes a 32x32 output tile (16 wmma/k-step
// from 8 fragment loads). TRANSPOSE_V==1 stores per-batch transposed [b][e][t].
// Block = 128 threads = 4 waves, each wave a different 32-wide n strip.
// ---------------------------------------------------------------------------
template <int TRANSPOSE_V>
__global__ __launch_bounds__(128) void proj_kernel(
    const _Float16* __restrict__ xr,  const _Float16* __restrict__ xi,
    const _Float16* __restrict__ wre, const _Float16* __restrict__ wim,
    const float* __restrict__ biasRe, const float* __restrict__ biasIm,
    _Float16* __restrict__ outRe, _Float16* __restrict__ outIm) {
  const int wave = threadIdx.x >> 5;
  const int lane = threadIdx.x & 31;
  const int r    = lane & 15;
  const int half = lane >> 4;
  const int m0   = blockIdx.x * 32;
  const int n0   = (blockIdx.y * 4 + wave) * 32;

  v8f accR[2][2] = {};
  v8f accI[2][2] = {};
#pragma unroll 4
  for (int kk = 0; kk < Dn; kk += 32) {
    v16h axr[2], axi[2], bre[2], bim[2];
#pragma unroll
    for (int i = 0; i < 2; ++i) {
      axr[i] = loadA(xr, Dn, m0 + 16 * i + r, kk, lane);
      axi[i] = loadA(xi, Dn, m0 + 16 * i + r, kk, lane);
    }
#pragma unroll
    for (int j = 0; j < 2; ++j) {
      bre[j] = loadB(wre, Dn, n0 + 16 * j + r, kk, lane);
      bim[j] = loadB(wim, Dn, n0 + 16 * j + r, kk, lane);
    }
#pragma unroll
    for (int j = 0; j < 2; ++j) {
      const v16h bimn = negH(bim[j]);
#pragma unroll
      for (int i = 0; i < 2; ++i) {
        accR[i][j] = wmma16(axr[i], bre[j], accR[i][j]);
        accR[i][j] = wmma16(axi[i], bimn,   accR[i][j]);
        accI[i][j] = wmma16(axr[i], bim[j], accI[i][j]);
        accI[i][j] = wmma16(axi[i], bre[j], accI[i][j]);
      }
    }
  }

#pragma unroll
  for (int j = 0; j < 2; ++j) {
    const int   e  = n0 + 16 * j + r;
    const float br = biasRe[e];
    const float bi = biasIm[e];
#pragma unroll
    for (int i = 0; i < 2; ++i) {
#pragma unroll
      for (int v = 0; v < 8; ++v) {
        const int   m  = m0 + 16 * i + v + 8 * half;
        const float vr = accR[i][j][v] + br;
        const float vi = accI[i][j][v] + bi;
        if (TRANSPOSE_V) {
          const int b = m >> 11;            // m / Sn
          const int t = m & (Sn - 1);
          const size_t idx = ((size_t)b * Dn + e) * Sn + t;
          outRe[idx] = (_Float16)vr;
          outIm[idx] = (_Float16)vi;
        } else {
          const size_t idx = (size_t)m * Dn + e;
          outRe[idx] = (_Float16)vr;
          outIm[idx] = (_Float16)vi;
        }
      }
    }
  }
}

// ---------------------------------------------------------------------------
// scores[b,s,t] = beta * (Qre[s,:].Kre[t,:] + Qim[s,:].Kim[t,:])
// 2x2 blocking: 8 wmma per k-step from 8 fragment loads.
// ---------------------------------------------------------------------------
__global__ __launch_bounds__(128) void scores_kernel(
    const _Float16* __restrict__ qre, const _Float16* __restrict__ qim,
    const _Float16* __restrict__ kre, const _Float16* __restrict__ kim,
    float* __restrict__ scores) {
  const int wave = threadIdx.x >> 5;
  const int lane = threadIdx.x & 31;
  const int r    = lane & 15;
  const int half = lane >> 4;
  const int b    = blockIdx.z;
  const int m0   = blockIdx.x * 32;                  // s tile base
  const int n0   = (blockIdx.y * 4 + wave) * 32;     // t tile base

  const size_t boff = (size_t)b * Sn * Dn;
  const _Float16* qre_b = qre + boff;
  const _Float16* qim_b = qim + boff;
  const _Float16* kre_b = kre + boff;
  const _Float16* kim_b = kim + boff;

  v8f acc[2][2] = {};
#pragma unroll 4
  for (int kk = 0; kk < Dn; kk += 32) {
    v16h aqr[2], aqi[2], bkr[2], bki[2];
#pragma unroll
    for (int i = 0; i < 2; ++i) {
      aqr[i] = loadA(qre_b, Dn, m0 + 16 * i + r, kk, lane);
      aqi[i] = loadA(qim_b, Dn, m0 + 16 * i + r, kk, lane);
    }
#pragma unroll
    for (int j = 0; j < 2; ++j) {
      bkr[j] = loadB(kre_b, Dn, n0 + 16 * j + r, kk, lane);
      bki[j] = loadB(kim_b, Dn, n0 + 16 * j + r, kk, lane);
    }
#pragma unroll
    for (int j = 0; j < 2; ++j)
#pragma unroll
      for (int i = 0; i < 2; ++i) {
        acc[i][j] = wmma16(aqr[i], bkr[j], acc[i][j]);
        acc[i][j] = wmma16(aqi[i], bki[j], acc[i][j]);
      }
  }

#pragma unroll
  for (int i = 0; i < 2; ++i)
#pragma unroll
    for (int j = 0; j < 2; ++j)
#pragma unroll
      for (int v = 0; v < 8; ++v) {
        const int s = m0 + 16 * i + v + 8 * half;
        const int t = n0 + 16 * j + r;
        scores[((size_t)b * Sn + s) * Sn + t] = BETA * acc[i][j][v];
      }
}

// ---------------------------------------------------------------------------
// Row softmax over S=2048; writes attn as f16. One 256-thread block per row.
// ---------------------------------------------------------------------------
__global__ __launch_bounds__(256) void softmax_kernel(
    const float* __restrict__ scores, _Float16* __restrict__ attn) {
  __shared__ float red[256];
  const int    tid = threadIdx.x;
  const size_t row = blockIdx.x;
  const float* sp = scores + row * Sn;
  _Float16*    ap = attn   + row * Sn;

  float v[8];
  float mx = -3.402823466e38f;
#pragma unroll
  for (int k = 0; k < 8; ++k) {
    v[k] = sp[k * 256 + tid];
    mx = fmaxf(mx, v[k]);
  }
  red[tid] = mx; __syncthreads();
  for (int s = 128; s > 0; s >>= 1) {
    if (tid < s) red[tid] = fmaxf(red[tid], red[tid + s]);
    __syncthreads();
  }
  mx = red[0]; __syncthreads();

  float sum = 0.0f;
#pragma unroll
  for (int k = 0; k < 8; ++k) {
    v[k] = __expf(v[k] - mx);
    sum += v[k];
  }
  red[tid] = sum; __syncthreads();
  for (int s = 128; s > 0; s >>= 1) {
    if (tid < s) red[tid] += red[tid + s];
    __syncthreads();
  }
  const float inv = 1.0f / red[0];
#pragma unroll
  for (int k = 0; k < 8; ++k)
    ap[k * 256 + tid] = (_Float16)(v[k] * inv);
}

// ---------------------------------------------------------------------------
// out[b,s,e] = sum_t attn[b,s,t] * (Vre + i*Vim)[b,t,e]; V stored transposed
// [b][e][t] so B-fragments read contiguous t. 2x2 blocking: 8 wmma per
// k-step from 6 fragment loads. Output complex64 interleaved.
// ---------------------------------------------------------------------------
__global__ __launch_bounds__(128) void pv_kernel(
    const _Float16* __restrict__ attn,
    const _Float16* __restrict__ vtre, const _Float16* __restrict__ vtim,
    float* __restrict__ out) {
  const int wave = threadIdx.x >> 5;
  const int lane = threadIdx.x & 31;
  const int r    = lane & 15;
  const int half = lane >> 4;
  const int b    = blockIdx.z;
  const int m0   = blockIdx.x * 32;                  // s tile base
  const int n0   = (blockIdx.y * 4 + wave) * 32;     // e tile base

  const _Float16* ab = attn + (size_t)b * Sn * Sn;   // ld = Sn
  const _Float16* vr = vtre + (size_t)b * Dn * Sn;   // ld = Sn, row = e
  const _Float16* vi = vtim + (size_t)b * Dn * Sn;

  v8f accR[2][2] = {};
  v8f accI[2][2] = {};
#pragma unroll 4
  for (int kk = 0; kk < Sn; kk += 32) {
    v16h pa[2], bvr[2], bvi[2];
#pragma unroll
    for (int i = 0; i < 2; ++i)
      pa[i] = loadA(ab, Sn, m0 + 16 * i + r, kk, lane);
#pragma unroll
    for (int j = 0; j < 2; ++j) {
      bvr[j] = loadB(vr, Sn, n0 + 16 * j + r, kk, lane);
      bvi[j] = loadB(vi, Sn, n0 + 16 * j + r, kk, lane);
    }
#pragma unroll
    for (int j = 0; j < 2; ++j)
#pragma unroll
      for (int i = 0; i < 2; ++i) {
        accR[i][j] = wmma16(pa[i], bvr[j], accR[i][j]);
        accI[i][j] = wmma16(pa[i], bvi[j], accI[i][j]);
      }
  }

#pragma unroll
  for (int i = 0; i < 2; ++i)
#pragma unroll
    for (int j = 0; j < 2; ++j)
#pragma unroll
      for (int v = 0; v < 8; ++v) {
        const int s = m0 + 16 * i + v + 8 * half;
        const int e = n0 + 16 * j + r;
        float2 val; val.x = accR[i][j][v]; val.y = accI[i][j][v];
        *reinterpret_cast<float2*>(out + (((size_t)b * Sn + s) * Dn + e) * 2) = val;
      }
}

// ---------------------------------------------------------------------------
extern "C" void kernel_launch(void* const* d_in, const int* in_sizes, int n_in,
                              void* d_out, int out_size, void* d_ws, size_t ws_size,
                              hipStream_t stream) {
  (void)in_sizes; (void)n_in; (void)out_size; (void)ws_size;

  const float* x_re  = (const float*)d_in[0];
  const float* x_im  = (const float*)d_in[1];
  const float* Wq_re = (const float*)d_in[2];
  const float* Wq_im = (const float*)d_in[3];
  const float* bq_re = (const float*)d_in[4];
  const float* bq_im = (const float*)d_in[5];
  const float* Wk_re = (const float*)d_in[6];
  const float* Wk_im = (const float*)d_in[7];
  const float* bk_re = (const float*)d_in[8];
  const float* bk_im = (const float*)d_in[9];
  const float* Wv_re = (const float*)d_in[10];
  const float* Wv_im = (const float*)d_in[11];
  const float* bv_re = (const float*)d_in[12];
  const float* bv_im = (const float*)d_in[13];
  float* out = (float*)d_out;  // complex64 interleaved (re, im)

  // Workspace layout (132 MB): attn (32 MB) aliases Q/K after they're consumed.
  const size_t MB = 1ull << 20;
  char* w = (char*)d_ws;
  _Float16* qre  = (_Float16*)(w + 0  * MB);
  _Float16* qim  = (_Float16*)(w + 8  * MB);
  _Float16* kre  = (_Float16*)(w + 16 * MB);
  _Float16* kim  = (_Float16*)(w + 24 * MB);
  _Float16* vtre = (_Float16*)(w + 32 * MB);   // [B][D][S]
  _Float16* vtim = (_Float16*)(w + 40 * MB);
  _Float16* xr16 = (_Float16*)(w + 48 * MB);
  _Float16* xi16 = (_Float16*)(w + 56 * MB);
  _Float16* w16  = (_Float16*)(w + 64 * MB);   // 6 x (512*512) f16 = 3 MB
  float*    sc   = (float*)   (w + 68 * MB);   // [B][S][S] f32 = 64 MB
  _Float16* attn = (_Float16*)(w + 0  * MB);   // [B][S][S] f16, aliases Q/K

  const int DD = Dn * Dn;
  _Float16* wqre = w16 + 0 * DD;  _Float16* wqim = w16 + 1 * DD;
  _Float16* wkre = w16 + 2 * DD;  _Float16* wkim = w16 + 3 * DD;
  _Float16* wvre = w16 + 4 * DD;  _Float16* wvim = w16 + 5 * DD;

  // 1) f32 -> f16 conversions
  {
    const int nx4 = (Mn * Dn) / 4;
    const int nw4 = DD / 4;
    cvt_f32_f16_kernel<<<(nx4 + 255) / 256, 256, 0, stream>>>(x_re,  xr16, nx4);
    cvt_f32_f16_kernel<<<(nx4 + 255) / 256, 256, 0, stream>>>(x_im,  xi16, nx4);
    cvt_f32_f16_kernel<<<(nw4 + 255) / 256, 256, 0, stream>>>(Wq_re, wqre, nw4);
    cvt_f32_f16_kernel<<<(nw4 + 255) / 256, 256, 0, stream>>>(Wq_im, wqim, nw4);
    cvt_f32_f16_kernel<<<(nw4 + 255) / 256, 256, 0, stream>>>(Wk_re, wkre, nw4);
    cvt_f32_f16_kernel<<<(nw4 + 255) / 256, 256, 0, stream>>>(Wk_im, wkim, nw4);
    cvt_f32_f16_kernel<<<(nw4 + 255) / 256, 256, 0, stream>>>(Wv_re, wvre, nw4);
    cvt_f32_f16_kernel<<<(nw4 + 255) / 256, 256, 0, stream>>>(Wv_im, wvim, nw4);
  }

  // 2) Complex projections (WMMA, 32x32 per wave).
  //    grid: 256 m-tiles x 4 n-groups (4 waves each -> 16 n-tiles of 32)
  {
    dim3 grid(Mn / 32, (Dn / 32) / 4);
    proj_kernel<0><<<grid, 128, 0, stream>>>(xr16, xi16, wqre, wqim, bq_re, bq_im, qre, qim);
    proj_kernel<0><<<grid, 128, 0, stream>>>(xr16, xi16, wkre, wkim, bk_re, bk_im, kre, kim);
    proj_kernel<1><<<grid, 128, 0, stream>>>(xr16, xi16, wvre, wvim, bv_re, bv_im, vtre, vtim);
  }

  // 3) scores = beta * Re(Q K^H)   (WMMA, f32 out)
  {
    dim3 grid(Sn / 32, (Sn / 32) / 4, Bn);
    scores_kernel<<<grid, 128, 0, stream>>>(qre, qim, kre, kim, sc);
  }

  // 4) row softmax -> attn f16 (overwrites Q/K region, which is now dead)
  softmax_kernel<<<Bn * Sn, 256, 0, stream>>>(sc, attn);

  // 5) out = attn @ V (complex, WMMA), interleaved complex64 store
  {
    dim3 grid(Sn / 32, (Dn / 32) / 4, Bn);
    pv_kernel<<<grid, 128, 0, stream>>>(attn, vtre, vtim, out);
  }
}
